// GAT_62182536511729
// MI455X (gfx1250) — compile-verified, matrix operand order checked by reading, not run on hardware
//
#include <hip/hip_runtime.h>
#include <hip/hip_bf16.h>

typedef __attribute__((ext_vector_type(16))) _Float16 v16h;
typedef __attribute__((ext_vector_type(8)))  _Float16 v8h;
typedef __attribute__((ext_vector_type(8)))  float    v8f;

#define GAT_N_DEFAULT 50000
#define GAT_HEADS 4
#define GAT_D 16   // HID == OUT == 16, so both layers have 64 output columns

// ---------------------------------------------------------------------------
// exact float atomic max via monotone integer encoding (native int atomics)
// ---------------------------------------------------------------------------
__device__ __forceinline__ void atomicMaxFloat(float* addr, float v) {
    int iv = __float_as_int(v);
    if (iv >= 0)
        atomicMax((int*)addr, iv);                    // non-negative floats: signed order
    else
        atomicMin((unsigned int*)addr, (unsigned)iv); // negative floats: reversed unsigned
}

__device__ __forceinline__ float fadd_atomic(float* p, float v) {
    return __hip_atomic_fetch_add(p, v, __ATOMIC_RELAXED, __HIP_MEMORY_SCOPE_AGENT);
}

__device__ __forceinline__ float lrelu02(float x) { return x > 0.f ? x : 0.2f * x; }

// ---------------------------------------------------------------------------
// fill u32 pattern (used to seed segment-max buffers with -inf)
// ---------------------------------------------------------------------------
__global__ void k_fill_u32(unsigned* __restrict__ p, unsigned v, int n) {
    int i = blockIdx.x * blockDim.x + threadIdx.x;
    if (i < n) p[i] = v;
}

// ---------------------------------------------------------------------------
// WMMA GEMM:  Hout[N,64] = op(X[N,KDIM]) @ Wm[64,KDIM]^T
// op = identity, or relu(x + preb[k]) when PREOP (fused layer-2 input prep).
// Block = 256 threads (8 waves). Each wave owns a 16-row tile; X and W are
// converted f32 -> f16 into LDS once, then consumed as 16x16x32 WMMA frags.
// ---------------------------------------------------------------------------
template <int KDIM, bool PREOP>
__global__ __launch_bounds__(256) void k_gemm(const float* __restrict__ X,
                                              const float* __restrict__ Wm,
                                              const float* __restrict__ preb,
                                              float* __restrict__ Hout,
                                              int nrows) {
    __shared__ __align__(16) _Float16 sW[64 * KDIM];
    __shared__ __align__(16) _Float16 sA[8][16 * KDIM];

    const int tid  = threadIdx.x;
    const int lane = tid & 31;
    const int wv   = tid >> 5;

    // stage W (64 x KDIM, row-major) as f16
    for (int i = tid; i < 64 * KDIM; i += 256) sW[i] = (_Float16)Wm[i];

    // stage this wave's 16-row A tile as f16 (with optional bias+relu pre-op)
    const int rowBase = blockIdx.x * 128 + wv * 16;
    for (int i = lane; i < 16 * KDIM; i += 32) {
        int r = i / KDIM, c = i - r * KDIM;
        int gr = rowBase + r;
        float v = 0.f;
        if (gr < nrows) {
            v = X[(size_t)gr * KDIM + c];
            if (PREOP) v = fmaxf(v + preb[c], 0.f);
        }
        sA[wv][i] = (_Float16)v;
    }
    __syncthreads();

    v8f acc[4];
    for (int nt = 0; nt < 4; ++nt)
        for (int j = 0; j < 8; ++j) acc[nt][j] = 0.f;

    const int r16  = lane & 15;
    const int hsel = (lane >> 4) * 8;   // K sub-range selector per half-wave

    for (int kt = 0; kt < KDIM / 32; ++kt) {
        // A fragment: lane = row (mod 16); v0..3 and v4..7 K-ranges per half
        const _Float16* pa = &sA[wv][r16 * KDIM + kt * 32];
        v16h a;
        {
            v8h lo = *(const v8h*)(pa + hsel);
            v8h hi = *(const v8h*)(pa + 16 + hsel);
            for (int j = 0; j < 8; ++j) { a[j] = lo[j]; a[j + 8] = hi[j]; }
        }
        for (int nt = 0; nt < 4; ++nt) {
            // B fragment: same lane pattern with output-column as the lane row
            const _Float16* pb = &sW[(nt * 16 + r16) * KDIM + kt * 32];
            v16h b;
            {
                v8h lo = *(const v8h*)(pb + hsel);
                v8h hi = *(const v8h*)(pb + 16 + hsel);
                for (int j = 0; j < 8; ++j) { b[j] = lo[j]; b[j + 8] = hi[j]; }
            }
            acc[nt] = __builtin_amdgcn_wmma_f32_16x16x32_f16(
                false, a, false, b, (short)0, acc[nt], false, false);
        }
    }

    // C/D layout: lane -> col = lane%16 ; VGPR j -> row = (lane/16)*8 + j
    const int rw = rowBase + (lane >> 4) * 8;
    for (int nt = 0; nt < 4; ++nt) {
        int col = nt * 16 + r16;
        for (int j = 0; j < 8; ++j) {
            int r = rw + j;
            if (r < nrows) Hout[(size_t)r * 64 + col] = acc[nt][j];
        }
    }
}

// ---------------------------------------------------------------------------
// per-(node,head) attention coefficients: el = <h[n,h,:], al[h,:]>, er likewise
// ---------------------------------------------------------------------------
__global__ void k_attn(const float* __restrict__ hmat, const float* __restrict__ al,
                       const float* __restrict__ ar, float* __restrict__ el,
                       float* __restrict__ er, int n) {
    int i = blockIdx.x * blockDim.x + threadIdx.x;
    if (i >= n * GAT_HEADS) return;
    int nd = i >> 2, h = i & 3;
    const float* hp = hmat + (size_t)nd * 64 + h * GAT_D;
    float a = 0.f, b = 0.f;
#pragma unroll
    for (int d = 0; d < GAT_D; ++d) {
        float v = hp[d];
        a += v * al[h * GAT_D + d];
        b += v * ar[h * GAT_D + d];
    }
    el[i] = a;
    er[i] = b;
}

// ---------------------------------------------------------------------------
// edge pass 1: per-dst running max of leaky_relu(el[src]+er[dst])
// ---------------------------------------------------------------------------
__global__ void k_edge_max(const int* __restrict__ src, const int* __restrict__ dst,
                           const float* __restrict__ el, const float* __restrict__ er,
                           float* __restrict__ m, int EH) {
    int i = blockIdx.x * blockDim.x + threadIdx.x;
    if (i >= EH) return;
    int e = i >> 2, h = i & 3;
    int s_ = src[e], d_ = dst[e];
    float x = lrelu02(el[s_ * 4 + h] + er[d_ * 4 + h]);
    atomicMaxFloat(&m[d_ * 4 + h], x);
}

// ---------------------------------------------------------------------------
// edge pass 2: per-dst sum of exp(e - m[dst])
// ---------------------------------------------------------------------------
__global__ void k_edge_sum(const int* __restrict__ src, const int* __restrict__ dst,
                           const float* __restrict__ el, const float* __restrict__ er,
                           const float* __restrict__ m, float* __restrict__ s, int EH) {
    int i = blockIdx.x * blockDim.x + threadIdx.x;
    if (i >= EH) return;
    int e = i >> 2, h = i & 3;
    int s_ = src[e], d_ = dst[e];
    float x = lrelu02(el[s_ * 4 + h] + er[d_ * 4 + h]);
    float a = expf(x - m[d_ * 4 + h]);
    fadd_atomic(&s[d_ * 4 + h], a);
}

// ---------------------------------------------------------------------------
// edge pass 3: agg[dst, h, d] += alpha * h[src, h, d]
// one thread per (edge, h, d); alpha recomputed (L2-resident scalars) to
// avoid a 12.8MB alpha buffer. Gathers of h[src] hit L2 (h is 12.8MB).
// ---------------------------------------------------------------------------
__global__ void k_edge_aggr(const int* __restrict__ src, const int* __restrict__ dst,
                            const float* __restrict__ el, const float* __restrict__ er,
                            const float* __restrict__ m, const float* __restrict__ s,
                            const float* __restrict__ hsrc, float* __restrict__ agg,
                            int E) {
    int i = blockIdx.x * blockDim.x + threadIdx.x;
    if (i >= E * 64) return;
    int e = i >> 6;
    int t = i & 63;       // t = h*16 + d
    int h = t >> 4;
    int s_ = src[e], d_ = dst[e];
    __builtin_prefetch(&hsrc[(size_t)s_ * 64 + t], 0, 0);   // global_prefetch_b8
    float x = lrelu02(el[s_ * 4 + h] + er[d_ * 4 + h]);
    float w = expf(x - m[d_ * 4 + h]) / s[d_ * 4 + h];
    fadd_atomic(&agg[(size_t)d_ * 64 + t], w * hsrc[(size_t)s_ * 64 + t]);
}

// ---------------------------------------------------------------------------
// final: head-mean of (agg2 + b2), then row log-softmax over 16 columns.
// wave32: lanes 0..15 = node A's columns, lanes 16..31 = node B's columns.
// ---------------------------------------------------------------------------
__global__ void k_final(const float* __restrict__ agg, const float* __restrict__ b2,
                        float* __restrict__ out, int n) {
    int tid = blockIdx.x * blockDim.x + threadIdx.x;
    int node = tid >> 4;
    int col  = tid & 15;
    if (node >= n) return;
    float y = 0.f;
#pragma unroll
    for (int h = 0; h < GAT_HEADS; ++h)
        y += agg[(size_t)node * 64 + h * 16 + col] + b2[h * 16 + col];
    y *= 0.25f;
    float mx = y;
    for (int o = 8; o > 0; o >>= 1) mx = fmaxf(mx, __shfl_xor(mx, o, 16));
    float ex = expf(y - mx);
    float sm = ex;
    for (int o = 8; o > 0; o >>= 1) sm += __shfl_xor(sm, o, 16);
    out[(size_t)node * 16 + col] = y - mx - logf(sm);
}

// ---------------------------------------------------------------------------
extern "C" void kernel_launch(void* const* d_in, const int* in_sizes, int n_in,
                              void* d_out, int out_size, void* d_ws, size_t ws_size,
                              hipStream_t stream) {
    const float* feat = (const float*)d_in[0];
    const int*   src  = (const int*)d_in[1];
    const int*   dst  = (const int*)d_in[2];
    const float* W1   = (const float*)d_in[3];
    const float* al1  = (const float*)d_in[4];
    const float* ar1  = (const float*)d_in[5];
    const float* b1   = (const float*)d_in[6];
    const float* W2   = (const float*)d_in[7];
    const float* al2  = (const float*)d_in[8];
    const float* ar2  = (const float*)d_in[9];
    const float* b2   = (const float*)d_in[10];
    float* out = (float*)d_out;

    const int N = in_sizes[0] / 128;
    const int E = in_sizes[1];

    // workspace layout (floats): 3 * N*64 node matrices + 4 * N*4 scalars
    float* ws   = (float*)d_ws;
    float* h1   = ws;                          // [N,64] layer-1 fc out; reused as agg2
    float* agg1 = h1 + (size_t)N * 64;         // [N,64] layer-1 aggregate
    float* h2   = agg1 + (size_t)N * 64;       // [N,64] layer-2 fc out
    float* el   = h2 + (size_t)N * 64;         // [N,4]
    float* er   = el + (size_t)N * 4;          // [N,4]
    float* mz   = er + (size_t)N * 4;          // [N,4] segment max
    float* sz   = mz + (size_t)N * 4;          // [N,4] segment sum
    float* agg2 = h1;                          // h1 is dead once gemm2 has run

    const int TB = 256;
    const int gN4  = (N * 4 + TB - 1) / TB;
    const int gE4  = (E * 4 + TB - 1) / TB;
    const int gE64 = (E * 64 + TB - 1) / TB;
    const int gGem = (N + 127) / 128;

    // ---- layer 1 ----
    hipMemsetAsync(agg1, 0, (size_t)N * 64 * sizeof(float), stream);
    hipMemsetAsync(sz, 0, (size_t)N * 4 * sizeof(float), stream);
    k_fill_u32<<<gN4, TB, 0, stream>>>((unsigned*)mz, 0xFF800000u, N * 4); // -inf

    k_gemm<128, false><<<gGem, TB, 0, stream>>>(feat, W1, nullptr, h1, N);
    k_attn<<<gN4, TB, 0, stream>>>(h1, al1, ar1, el, er, N);
    k_edge_max<<<gE4, TB, 0, stream>>>(src, dst, el, er, mz, E * 4);
    k_edge_sum<<<gE4, TB, 0, stream>>>(src, dst, el, er, mz, sz, E * 4);
    k_edge_aggr<<<gE64, TB, 0, stream>>>(src, dst, el, er, mz, sz, h1, agg1, E);

    // ---- layer 2 (bias + relu fused into GEMM A-load) ----
    k_gemm<64, true><<<gGem, TB, 0, stream>>>(agg1, W2, b1, h2, N);
    k_attn<<<gN4, TB, 0, stream>>>(h2, al2, ar2, el, er, N);

    hipMemsetAsync(sz, 0, (size_t)N * 4 * sizeof(float), stream);
    k_fill_u32<<<gN4, TB, 0, stream>>>((unsigned*)mz, 0xFF800000u, N * 4);
    hipMemsetAsync(agg2, 0, (size_t)N * 64 * sizeof(float), stream);

    k_edge_max<<<gE4, TB, 0, stream>>>(src, dst, el, er, mz, E * 4);
    k_edge_sum<<<gE4, TB, 0, stream>>>(src, dst, el, er, mz, sz, E * 4);
    k_edge_aggr<<<gE64, TB, 0, stream>>>(src, dst, el, er, mz, sz, h2, agg2, E);

    // ---- head mean + log_softmax ----
    k_final<<<(N * 16 + TB - 1) / TB, TB, 0, stream>>>(agg2, b2, out, N);
}
